// DenormProject_39771397161210
// MI455X (gfx1250) — matrix operand order repaired and checked by reading, not compiled, not run on hardware
//
#include <hip/hip_runtime.h>
#include <hip/hip_bf16.h>

typedef __attribute__((ext_vector_type(16))) _Float16 v16h;
typedef __attribute__((ext_vector_type(8)))  _Float16 v8h;
typedef __attribute__((ext_vector_type(8)))  float    v8f;

#define B_      4
#define V_      5
#define NBINS_  128000
#define IMG_W_  960
#define IMG_H_  512
#define HM_W_   240
#define HM_H_   128
#define C_IMG_  3
#define C_FEAT_ 32
#define C_TOT_  35
#define HID_    256
#define DIN_    73
#define K1_     96          // D_IN padded 73 -> 96 (3 x K=32 WMMA steps)
#define MTILE   64          // points per block
#define XS_STRIDE 104       // halves per Xs row (208 B, 16B aligned)
#define H1_STRIDE 264       // halves per H1 row (528 B, 16B aligned)

// ---------------- weight prep: f32 -> transposed f16 ----------------
// Wt1[n][k] = (k<73) ? W1[k][n] : 0     (256 x 96)
// Wt2[n][k] = W2[k][n]                  (256 x 256)
// W3f[n]    = W3[n][0]                  (256)
__global__ void prep_weights(const float* __restrict__ W1,
                             const float* __restrict__ W2,
                             const float* __restrict__ W3,
                             _Float16* __restrict__ Wt1,
                             _Float16* __restrict__ Wt2,
                             float* __restrict__ W3f) {
  int idx = blockIdx.x * blockDim.x + threadIdx.x;
  if (idx < HID_ * K1_) {
    int n = idx / K1_, k = idx % K1_;
    Wt1[idx] = (k < DIN_) ? (_Float16)W1[k * HID_ + n] : (_Float16)0.0f;
  }
  int j = idx - HID_ * K1_;
  if (j >= 0 && j < HID_ * HID_) {
    int n = j / HID_, k = j % HID_;
    Wt2[j] = (_Float16)W2[k * HID_ + n];
  }
  int m = idx - HID_ * K1_ - HID_ * HID_;
  if (m >= 0 && m < HID_) W3f[m] = W3[m];
}

// ---------------- bilinear sampler (align_corners=True, zero pad) ---
template <int C>
__device__ __forceinline__ void sample_add(const float* __restrict__ img,
                                           int H, int W, float gx, float gy,
                                           float* __restrict__ s,
                                           float* __restrict__ q) {
  float x = (gx + 1.0f) * 0.5f * (float)(W - 1);
  float y = (gy + 1.0f) * 0.5f * (float)(H - 1);
  float x0f = floorf(x), y0f = floorf(y);
  int x0 = (int)x0f, y0 = (int)y0f;
  float wx1 = x - x0f, wy1 = y - y0f;
  float wx0 = 1.0f - wx1, wy0 = 1.0f - wy1;
  int x1 = x0 + 1, y1 = y0 + 1;
  bool vx0 = (x0 >= 0) && (x0 < W), vx1 = (x1 >= 0) && (x1 < W);
  bool vy0 = (y0 >= 0) && (y0 < H), vy1 = (y1 >= 0) && (y1 < H);
  int cx0 = x0 < 0 ? 0 : (x0 > W - 1 ? W - 1 : x0);
  int cx1 = x1 < 0 ? 0 : (x1 > W - 1 ? W - 1 : x1);
  int cy0 = y0 < 0 ? 0 : (y0 > H - 1 ? H - 1 : y0);
  int cy1 = y1 < 0 ? 0 : (y1 > H - 1 ? H - 1 : y1);
  float m00 = (vx0 && vy0) ? wx0 * wy0 : 0.0f;
  float m10 = (vx1 && vy0) ? wx1 * wy0 : 0.0f;
  float m01 = (vx0 && vy1) ? wx0 * wy1 : 0.0f;
  float m11 = (vx1 && vy1) ? wx1 * wy1 : 0.0f;
  size_t hw = (size_t)H * (size_t)W;
  size_t o00 = (size_t)cy0 * W + cx0, o10 = (size_t)cy0 * W + cx1;
  size_t o01 = (size_t)cy1 * W + cx0, o11 = (size_t)cy1 * W + cx1;
#pragma unroll
  for (int ch = 0; ch < C; ++ch) {
    const float* p = img + (size_t)ch * hw;
    float v = p[o00] * m00 + p[o10] * m10 + p[o01] * m01 + p[o11] * m11;
    s[ch] += v;
    q[ch] += v * v;
  }
}

// ---------------- fused sampling + MLP (WMMA) -----------------------
__global__ __launch_bounds__(128) void voxelpose_fused_kernel(
    const float* __restrict__ imgs,    // [V,B,3,512,960]
    const float* __restrict__ feats,   // [V,B,32,128,240]
    const float* __restrict__ grids,   // [B,N,3]
    const float* __restrict__ Rm,      // [V,B,3,3]
    const float* __restrict__ Tm,      // [V,B,3,1]
    const float* __restrict__ fm,      // [V,B,2]
    const float* __restrict__ cm,      // [V,B,2]
    const float* __restrict__ kd,      // [V,B,3]
    const float* __restrict__ pd,      // [V,B,2]
    const float* __restrict__ trans,   // [2,3]
    const _Float16* __restrict__ Wt1,  // [256,96]  (N-major, K-contig)
    const float* __restrict__ b1,
    const _Float16* __restrict__ Wt2,  // [256,256]
    const float* __restrict__ b2,
    const float* __restrict__ W3f,     // [256]
    const float* __restrict__ b3,
    float* __restrict__ out) {
  __shared__ _Float16 Xs[MTILE * XS_STRIDE];   // 13,312 B activation tile (f16)
  __shared__ _Float16 H1s[MTILE * H1_STRIDE];  // 33,792 B hidden tile (f16)
  __shared__ float red[MTILE * 16];            //  4,096 B layer3 reduction
  __shared__ float cvec[3 * HID_];             //  3,072 B b1|b2|W3 (async DMA)
  float* comb = (float*)H1s;                   // overlay: view-combine scratch

  const int tid = threadIdx.x;
  const int lane = tid & 31;
  const int wave = tid >> 5;
  const int p = tid & 63;     // point within tile
  const int grp = tid >> 6;   // 0: views 0..2, 1: views 3..4
  const long p0 = (long)blockIdx.x * MTILE;
  const long pr = p0 + p;     // global point row (== flat out index)
  const int b = (int)(p0 / NBINS_);

  // ---- phase 0: kick off async global->LDS DMA of b1|b2|W3 (3 KB). ----
  // Copies run concurrently with the whole sampling phase; each wave
  // waits on its own ASYNCcnt before the pre-WMMA barrier.
#pragma unroll
  for (int i = tid; i < 192; i += 128) {
    const float* src = (i < 64)    ? (b1 + i * 4)
                       : (i < 128) ? (b2 + (i - 64) * 4)
                                   : (W3f + (i - 128) * 4);
    unsigned ldsa = (unsigned)(uintptr_t)(&cvec[i * 4]);  // LDS offset bits
    asm volatile("global_load_async_to_lds_b128 %0, %1, off"
                 :: "v"(ldsa), "v"(src)
                 : "memory");
  }
  __builtin_prefetch((const void*)Wt1, 0, 1);
  __builtin_prefetch((const void*)Wt2, 0, 1);

  // ---- phase 1: project + bilinear sample, per-view accumulation ----
  float g3[3];
  g3[0] = grids[pr * 3 + 0];
  g3[1] = grids[pr * 3 + 1];
  g3[2] = grids[pr * 3 + 2];

  float sums[C_TOT_], sqs[C_TOT_];
#pragma unroll
  for (int i = 0; i < C_TOT_; ++i) { sums[i] = 0.0f; sqs[i] = 0.0f; }

  const float t00 = trans[0], t01 = trans[1], t02 = trans[2];
  const float t10 = trans[3], t11 = trans[4], t12 = trans[5];

  const int v_lo = (grp == 0) ? 0 : 3;
  const int v_hi = (grp == 0) ? 3 : 5;
  for (int v = v_lo; v < v_hi; ++v) {
    const int vb = v * B_ + b;
    const float* R9 = Rm + vb * 9;
    const float* T3 = Tm + vb * 3;
    float dx = g3[0] - T3[0], dy = g3[1] - T3[1], dz = g3[2] - T3[2];
    float xc = R9[0] * dx + R9[1] * dy + R9[2] * dz;
    float yc = R9[3] * dx + R9[4] * dy + R9[5] * dz;
    float zc = R9[6] * dx + R9[7] * dy + R9[8] * dz;
    float inv = 1.0f / (zc + 1e-5f);
    float y0 = xc * inv, y1 = yc * inv;
    float r2 = y0 * y0 + y1 * y1;
    const float* kk = kd + vb * 3;
    const float* pp = pd + vb * 2;
    float radial = 1.0f + kk[0] * r2 + kk[1] * r2 * r2 + kk[2] * r2 * r2 * r2;
    float tn = pp[0] * y1 + pp[1] * y0;
    float u0 = y0 * (radial + 2.0f * tn) + pp[1] * r2;
    float u1 = y1 * (radial + 2.0f * tn) + pp[0] * r2;
    float px = fm[vb * 2 + 0] * u0 + cm[vb * 2 + 0];
    float py = fm[vb * 2 + 1] * u1 + cm[vb * 2 + 1];
    px = fminf(fmaxf(px, -1.0f), 1920.0f);
    py = fminf(fmaxf(py, -1.0f), 1920.0f);
    float xr = t00 * px + t01 * py + t02;
    float yr = t10 * px + t11 * py + t12;
    float gx = xr / (IMG_W_ - 1.0f) * 2.0f - 1.0f;
    float gy = yr / (IMG_H_ - 1.0f) * 2.0f - 1.0f;
    sample_add<C_IMG_>(imgs + (size_t)vb * C_IMG_ * IMG_H_ * IMG_W_,
                       IMG_H_, IMG_W_, gx, gy, sums, sqs);
    sample_add<C_FEAT_>(feats + (size_t)vb * C_FEAT_ * HM_H_ * HM_W_,
                        HM_H_, HM_W_, gx, gy, sums + C_IMG_, sqs + C_IMG_);
  }

  // ---- phase 2: combine two view-groups, build f16 activation row ----
  if (grp == 1) {
#pragma unroll
    for (int ch = 0; ch < C_TOT_; ++ch) {
      comb[p * C_TOT_ + ch] = sums[ch];
      comb[MTILE * C_TOT_ + p * C_TOT_ + ch] = sqs[ch];
    }
  }
  __syncthreads();
  if (grp == 0) {
    _Float16* xrow = &Xs[p * XS_STRIDE];
    xrow[0] = (_Float16)(g3[0] * 1e-3f);
    xrow[1] = (_Float16)(g3[1] * 1e-3f);
    xrow[2] = (_Float16)(g3[2] * 1e-3f);
#pragma unroll
    for (int ch = 0; ch < C_TOT_; ++ch) {
      float tot = sums[ch] + comb[p * C_TOT_ + ch];
      float tsq = sqs[ch] + comb[MTILE * C_TOT_ + p * C_TOT_ + ch];
      float mean = tot * 0.2f;
      float var = tsq * 0.2f - mean * mean;
      var = fmaxf(var, 0.0f);
      xrow[3 + ch] = (_Float16)mean;
      xrow[3 + C_TOT_ + ch] = (_Float16)__expf(-var);
    }
#pragma unroll
    for (int j = 3 + 2 * C_TOT_; j < K1_; ++j) xrow[j] = (_Float16)0.0f;
  }
  // each wave drains its own async copies, then make them block-visible
  asm volatile("s_wait_asynccnt 0x0" ::: "memory");
  __syncthreads();

  // ---- phase 3: WMMA MLP. wave w owns rows [w*16, w*16+16) ----------
  const int hs = lane >> 4;     // half-wave select
  const int lcol = lane & 15;
  const int wrow0 = wave * 16;

  // Layer 1: H1 = relu(X[64x96] @ W1[96x256] + b1)
  for (int nt = 0; nt < 16; ++nt) {
    const int n0 = nt * 16;
    float bias = cvec[n0 + lcol];
    v8f acc = {bias, bias, bias, bias, bias, bias, bias, bias};
#pragma unroll
    for (int kt = 0; kt < 3; ++kt) {
      const int k0 = kt * 32;
      // A fragment (16x32 f16): lanes 0-15 -> K {0-7,16-23}, 16-31 -> K {8-15,24-31}
      const _Float16* pA = &Xs[(wrow0 + lcol) * XS_STRIDE + k0 + hs * 8];
      v8h alo = *(const v8h*)pA;
      v8h ahi = *(const v8h*)(pA + 16);
      v16h a;
#pragma unroll
      for (int i = 0; i < 8; ++i) { a[i] = alo[i]; a[8 + i] = ahi[i]; }
      // B fragment (32x16 f16): lane = column, K-contiguous in transposed weights
      const _Float16* pB = Wt1 + (size_t)(n0 + lcol) * K1_ + k0 + hs * 8;
      v8h blo = *(const v8h*)pB;
      v8h bhi = *(const v8h*)(pB + 16);
      v16h bf;
#pragma unroll
      for (int i = 0; i < 8; ++i) { bf[i] = blo[i]; bf[8 + i] = bhi[i]; }
      acc = __builtin_amdgcn_wmma_f32_16x16x32_f16(false, a, false, bf,
                                                   (short)0, acc, false, false);
    }
#pragma unroll
    for (int i = 0; i < 8; ++i) {
      int rm = hs * 8 + i;  // D layout: VGPR i -> row i (lanes<16) / 8+i
      H1s[(wrow0 + rm) * H1_STRIDE + n0 + lcol] = (_Float16)fmaxf(acc[i], 0.0f);
    }
  }
  __syncthreads();

  // Layer 2 (+ fused layer 3): H2 = relu(H1 @ W2 + b2); part += H2 * W3
  float part[8];
#pragma unroll
  for (int i = 0; i < 8; ++i) part[i] = 0.0f;
  for (int nt = 0; nt < 16; ++nt) {
    const int n0 = nt * 16;
    float bias = cvec[HID_ + n0 + lcol];
    v8f acc = {bias, bias, bias, bias, bias, bias, bias, bias};
#pragma unroll
    for (int kt = 0; kt < 8; ++kt) {
      const int k0 = kt * 32;
      const _Float16* pA = &H1s[(wrow0 + lcol) * H1_STRIDE + k0 + hs * 8];
      v8h alo = *(const v8h*)pA;
      v8h ahi = *(const v8h*)(pA + 16);
      v16h a;
#pragma unroll
      for (int i = 0; i < 8; ++i) { a[i] = alo[i]; a[8 + i] = ahi[i]; }
      const _Float16* pB = Wt2 + (size_t)(n0 + lcol) * HID_ + k0 + hs * 8;
      v8h blo = *(const v8h*)pB;
      v8h bhi = *(const v8h*)(pB + 16);
      v16h bf;
#pragma unroll
      for (int i = 0; i < 8; ++i) { bf[i] = blo[i]; bf[8 + i] = bhi[i]; }
      acc = __builtin_amdgcn_wmma_f32_16x16x32_f16(false, a, false, bf,
                                                   (short)0, acc, false, false);
    }
    float w3 = cvec[2 * HID_ + n0 + lcol];
#pragma unroll
    for (int i = 0; i < 8; ++i) part[i] += fmaxf(acc[i], 0.0f) * w3;
  }

  // ---- phase 4: cross-lane reduction + softplus/alpha epilogue ------
#pragma unroll
  for (int i = 0; i < 8; ++i) {
    int rm = hs * 8 + i;
    red[(wrow0 + rm) * 16 + lcol] = part[i];
  }
  __syncthreads();
  if (tid < MTILE) {
    float s = 0.0f;
#pragma unroll
    for (int j = 0; j < 16; ++j) s += red[tid * 16 + j];
    s += b3[0];
    float sp = fmaxf(s, 0.0f) + log1pf(__expf(-fabsf(s)));  // softplus
    out[p0 + tid] = 1.0f - __expf(-sp);                      // alpha
  }
}

// ---------------- host launcher ------------------------------------
extern "C" void kernel_launch(void* const* d_in, const int* in_sizes, int n_in,
                              void* d_out, int out_size, void* d_ws,
                              size_t ws_size, hipStream_t stream) {
  const float* imgs  = (const float*)d_in[0];
  const float* feats = (const float*)d_in[1];
  const float* grids = (const float*)d_in[2];
  const float* Rm    = (const float*)d_in[3];
  const float* Tm    = (const float*)d_in[4];
  const float* fm    = (const float*)d_in[5];
  const float* cm    = (const float*)d_in[6];
  const float* kd    = (const float*)d_in[7];
  const float* pd    = (const float*)d_in[8];
  const float* trans = (const float*)d_in[9];
  const float* W1    = (const float*)d_in[10];
  const float* b1    = (const float*)d_in[11];
  const float* W2    = (const float*)d_in[12];
  const float* b2    = (const float*)d_in[13];
  const float* W3    = (const float*)d_in[14];
  const float* b3    = (const float*)d_in[15];
  float* out = (float*)d_out;

  char* ws = (char*)d_ws;
  _Float16* Wt1 = (_Float16*)(ws);                       // 49,152 B
  _Float16* Wt2 = (_Float16*)(ws + 49152);               // 131,072 B
  float* W3f    = (float*)(ws + 49152 + 131072);         // 1,024 B

  const int prep_elems = HID_ * K1_ + HID_ * HID_ + HID_;
  prep_weights<<<(prep_elems + 255) / 256, 256, 0, stream>>>(W1, W2, W3, Wt1,
                                                             Wt2, W3f);

  const int nblocks = (B_ * NBINS_) / MTILE;  // 8000
  voxelpose_fused_kernel<<<nblocks, 128, 0, stream>>>(
      imgs, feats, grids, Rm, Tm, fm, cm, kd, pd, trans, Wt1, b1, Wt2, b2, W3f,
      b3, out);
}
